// FaceModel_56444460204058
// MI455X (gfx1250) — compile-verified, matrix operand order checked by reading, not run on hardware
//
#include <hip/hip_runtime.h>

// ---------------------------------------------------------------------------
// StyleGAN2 modulated conv (B=8, Cin=Cout=512, 3x3, 64x64) for gfx1250.
// Fold style into x (bf16 NHWC), conv with batch-independent bf16 weights via
// implicit GEMM (v_wmma_f32_16x16x32_bf16), fold SCALE*demod + noise + bias +
// leaky_relu*sqrt(2) into the epilogue.
//
// Main kernel: 256 thr (8 waves), block tile 128co x 128px, wave tile 32x64.
// K-loop = 9 taps x 16 ci-chunks of 32 = 144 steps. Triple-buffered LDS fed
// by GLOBAL_LOAD_ASYNC_TO_LDS_B128 (ASYNCcnt), one barrier per step, halo
// rows redirected to a zeroed 128B workspace page (uniform control flow).
//
// Workspace (bytes): xs bf16 NHWC 32MB | wq bf16 [9][Cout][Cin] 4.5MB |
//                    s2 f32 1MB | epi f32 16KB | zpad 128B  (~37.6MB total)
// ---------------------------------------------------------------------------

typedef __attribute__((ext_vector_type(8)))  float  v8f;
typedef __attribute__((ext_vector_type(8)))  __bf16 v8bf;
typedef __attribute__((ext_vector_type(16))) __bf16 v16bf;

#define CIN   512
#define COUT  512
#define IMH   64
#define IMW   64
#define NB    8
#define NPIX  (NB * IMH * IMW)          // 32768
#define NSTEP 144                       // 9 taps * 16 ci chunks
#define LROW  40                        // LDS row stride in bf16 elements
#define SCALE_F 0.014731391274719738f   // 1/sqrt(512*9)
#define LRELU_F 1.4142135623730951f

#define WS_XS   0
#define WS_WQ   33554432u
#define WS_S2   38273024u
#define WS_EPI  39321600u
#define WS_ZPAD 39337984u

__device__ __forceinline__ v16bf cat16(v8bf lo, v8bf hi) {
  return __builtin_shufflevector(lo, hi, 0,1,2,3,4,5,6,7,8,9,10,11,12,13,14,15);
}
__device__ __forceinline__ v8bf ldv8(const __bf16* p) { return *(const v8bf*)p; }

// async global->LDS 32B (two b128, INST_OFFSET applies to both sides)
__device__ __forceinline__ void async_cp32(unsigned ldsoff, const __bf16* g) {
  asm volatile("global_load_async_to_lds_b128 %0, %1, off"
               :: "v"(ldsoff), "v"(g) : "memory");
  asm volatile("global_load_async_to_lds_b128 %0, %1, off offset:16"
               :: "v"(ldsoff), "v"(g) : "memory");
}

// ---- xs[b][h][w][ci] = bf16(style[b][ci] * x[b][ci][h][w]) -----------------
__global__ __launch_bounds__(256) void k_modx(const float* __restrict__ x,
                                              const float* __restrict__ style,
                                              __bf16* __restrict__ xs) {
  __shared__ __align__(16) __bf16 tile[64 * 65];
  const int bh = blockIdx.x;              // B*H blocks
  const int b  = bh >> 6, h = bh & 63;
  const int t  = threadIdx.x;
  for (int c0 = 0; c0 < CIN; c0 += 64) {
    #pragma unroll 4
    for (int it = 0; it < 16; ++it) {     // read coalesced along w
      int idx = it * 256 + t;
      int ci = idx >> 6, w = idx & 63;
      float s = style[b * CIN + c0 + ci];
      float v = x[(((size_t)(b * CIN + c0 + ci) * IMH) + h) * IMW + w];
      tile[ci * 65 + w] = (__bf16)(s * v);
    }
    __syncthreads();
    #pragma unroll 4
    for (int it = 0; it < 16; ++it) {     // write coalesced along ci
      int idx = it * 256 + t;
      int w = idx >> 6, ci = idx & 63;
      xs[((size_t)(b * 4096 + h * 64 + w) << 9) + c0 + ci] = tile[ci * 65 + w];
    }
    __syncthreads();
  }
}

// ---- wq[kk][co][ci] = bf16(weight[co][ci][kk]);  s2[co][ci] = sum_k w^2 ----
// also zero-fills the 128B halo zero-page
__global__ __launch_bounds__(512) void k_wprep(const float* __restrict__ weight,
                                               __bf16* __restrict__ wq,
                                               float* __restrict__ s2,
                                               float* __restrict__ zpad) {
  if (blockIdx.x == 0 && threadIdx.x < 32) zpad[threadIdx.x] = 0.f;
  const int co = blockIdx.x, ci = threadIdx.x;
  const float* wp = weight + (size_t)(co * CIN + ci) * 9;
  float s = 0.f, wv[9];
  #pragma unroll
  for (int k = 0; k < 9; ++k) { wv[k] = wp[k]; s += wv[k] * wv[k]; }
  s2[co * CIN + ci] = s;
  #pragma unroll
  for (int k = 0; k < 9; ++k)
    wq[(((size_t)k * COUT + co) << 9) + ci] = (__bf16)wv[k];
}

// ---- epi[b][co] = SCALE * rsqrt(SCALE^2 * sum_ci s2*style^2 + 1e-8) --------
__global__ __launch_bounds__(256) void k_demod(const float* __restrict__ style,
                                               const float* __restrict__ s2,
                                               float* __restrict__ epi) {
  const int co = blockIdx.x;
  const int lane = threadIdx.x & 31, b = threadIdx.x >> 5;   // wave per batch
  float sum = 0.f;
  for (int ci = lane; ci < CIN; ci += 32) {
    float st = style[b * CIN + ci];
    sum += s2[co * CIN + ci] * st * st;
  }
  #pragma unroll
  for (int off = 16; off > 0; off >>= 1) sum += __shfl_down(sum, off, 32);
  if (lane == 0)
    epi[(b << 9) + co] = SCALE_F * rsqrtf(SCALE_F * SCALE_F * sum + 1e-8f);
}

// ---- main implicit-GEMM conv ----------------------------------------------
__global__ __launch_bounds__(256) void k_conv(const __bf16* __restrict__ xs,
                                              const __bf16* __restrict__ wq,
                                              const float*  __restrict__ epi,
                                              const float*  __restrict__ noise,
                                              const float*  __restrict__ bias,
                                              const float*  __restrict__ nstr,
                                              const __bf16* __restrict__ zpad,
                                              float* __restrict__ out) {
  // [buf][A=0/Bt=1][128 rows * 40] bf16 : 60 KB, triple-buffered
  __shared__ __align__(16) __bf16 sh[3][2][128 * LROW];

  const int t       = threadIdx.x;
  const int px_base = blockIdx.x * 128;
  const int co_blk  = blockIdx.y * 128;
  const int wave = t >> 5, lane = t & 31;
  const int wco = wave >> 1, wpx = wave & 1;      // 4x2 wave grid, tile 32x64
  const int g = lane >> 4, l15 = lane & 15;

  // staging role: row = t>>1 (0..127), half = t&1 (32B per thread per tile)
  const int srow = t >> 1, shalf = t & 1;
  const int pxn = px_base + srow;
  const int bb = pxn >> 12, hh = (pxn >> 6) & 63, ww = pxn & 63;
  const long xbase = ((long)(bb << 12) + hh * 64 + ww) << 9;

  // 9-bit tap validity mask for this thread's staged pixel row
  unsigned vmask = 0;
  #pragma unroll
  for (int kk = 0; kk < 9; ++kk) {
    int h2 = hh + kk / 3 - 1, w2 = ww + kk % 3 - 1;
    if ((unsigned)h2 < (unsigned)IMH && (unsigned)w2 < (unsigned)IMW)
      vmask |= 1u << kk;
  }

  // LDS addresses per buffer (async wants raw 32-bit LDS byte offsets)
  unsigned stA[3], stB[3];
  const __bf16 *ldA[3], *ldB[3];
  #pragma unroll
  for (int p = 0; p < 3; ++p) {
    stA[p] = (unsigned)(uintptr_t)&sh[p][0][srow * LROW + shalf * 16];
    stB[p] = (unsigned)(uintptr_t)&sh[p][1][srow * LROW + shalf * 16];
    ldA[p] = &sh[p][0][(wco * 32 + l15) * LROW + g * 8];
    ldB[p] = &sh[p][1][(wpx * 64 + l15) * LROW + g * 8];
  }

  v8f acc[2][4] = {};

  auto issue = [&](int p, int itn) {              // async-stage step itn
    int kk  = itn >> 4;
    int ci0 = (itn & 15) << 5;
    const __bf16* ga = wq + ((((size_t)kk * COUT) + co_blk + srow) << 9)
                          + ci0 + shalf * 16;
    async_cp32(stA[p], ga);
    int dh = kk / 3, dw = kk - dh * 3;
    long toff = (long)(((dh - 1) * 64 + (dw - 1)) << 9);
    const __bf16* gb = xs + xbase + toff + ci0 + shalf * 16;
    if (!((vmask >> kk) & 1u)) gb = zpad;         // halo -> zero page
    async_cp32(stB[p], gb);
  };

  auto body = [&](int p, int it) {
    // oldest 4 asyncs (this step's tile) must have landed; newest 4 in flight
    asm volatile("s_wait_asynccnt 0x4" ::: "memory");
    __syncthreads();
    // stage step it+2 into buffer (p+2)%3 (safe: readers of that buffer
    // drained before they signalled this barrier)
    int nxt = it + 2;
    issue((p + 2) % 3, nxt < NSTEP ? nxt : it);
    // fragments: lane=16g+m, elems[0..7]=K 8g.., [8..15]=K 16+8g..
    v16bf a0 = cat16(ldv8(ldA[p]),            ldv8(ldA[p] + 16));
    v16bf a1 = cat16(ldv8(ldA[p] + 16*LROW),  ldv8(ldA[p] + 16*LROW + 16));
    v16bf b0 = cat16(ldv8(ldB[p]),            ldv8(ldB[p] + 16));
    v16bf b1 = cat16(ldv8(ldB[p] + 16*LROW),  ldv8(ldB[p] + 16*LROW + 16));
    v16bf b2 = cat16(ldv8(ldB[p] + 32*LROW),  ldv8(ldB[p] + 32*LROW + 16));
    v16bf b3 = cat16(ldv8(ldB[p] + 48*LROW),  ldv8(ldB[p] + 48*LROW + 16));
    acc[0][0] = __builtin_amdgcn_wmma_f32_16x16x32_bf16(false, a0, false, b0,
                                                (short)0, acc[0][0], false, false);
    acc[0][1] = __builtin_amdgcn_wmma_f32_16x16x32_bf16(false, a0, false, b1,
                                                (short)0, acc[0][1], false, false);
    acc[0][2] = __builtin_amdgcn_wmma_f32_16x16x32_bf16(false, a0, false, b2,
                                                (short)0, acc[0][2], false, false);
    acc[0][3] = __builtin_amdgcn_wmma_f32_16x16x32_bf16(false, a0, false, b3,
                                                (short)0, acc[0][3], false, false);
    acc[1][0] = __builtin_amdgcn_wmma_f32_16x16x32_bf16(false, a1, false, b0,
                                                (short)0, acc[1][0], false, false);
    acc[1][1] = __builtin_amdgcn_wmma_f32_16x16x32_bf16(false, a1, false, b1,
                                                (short)0, acc[1][1], false, false);
    acc[1][2] = __builtin_amdgcn_wmma_f32_16x16x32_bf16(false, a1, false, b2,
                                                (short)0, acc[1][2], false, false);
    acc[1][3] = __builtin_amdgcn_wmma_f32_16x16x32_bf16(false, a1, false, b3,
                                                (short)0, acc[1][3], false, false);
  };

  issue(0, 0);                                    // prologue: steps 0 and 1
  issue(1, 1);
  #pragma unroll 1
  for (int it = 0; it < NSTEP; it += 3) {         // buffer-index unrolled x3
    body(0, it);
    body(1, it + 1);
    body(2, it + 2);
  }

  // epilogue: y = acc*epi + ns*noise + bias; leaky_relu(0.2)*sqrt(2)
  const float ns = nstr[0];
  #pragma unroll
  for (int i = 0; i < 2; ++i) {
    #pragma unroll
    for (int j = 0; j < 4; ++j) {
      int px = px_base + wpx * 64 + j * 16 + l15;   // N = lane&15
      int b_ = px >> 12, hw = px & 4095;
      float nz = ns * noise[hw];
      int co0 = co_blk + wco * 32 + i * 16 + g * 8; // M = r + 8g
      #pragma unroll
      for (int r = 0; r < 8; ++r) {
        int co = co0 + r;
        float y = acc[i][j][r] * epi[(b_ << 9) + co] + nz + bias[co];
        y = (y >= 0.f ? y : 0.2f * y) * LRELU_F;
        out[(((size_t)(b_ << 9) + co) << 12) + hw] = y;
      }
    }
  }
}

extern "C" void kernel_launch(void* const* d_in, const int* in_sizes, int n_in,
                              void* d_out, int out_size, void* d_ws, size_t ws_size,
                              hipStream_t stream) {
  (void)in_sizes; (void)n_in; (void)out_size; (void)ws_size;
  const float* x      = (const float*)d_in[0];
  const float* style  = (const float*)d_in[1];
  const float* noise  = (const float*)d_in[2];
  const float* weight = (const float*)d_in[3];
  const float* bias   = (const float*)d_in[4];
  const float* nstr   = (const float*)d_in[5];
  float* out = (float*)d_out;

  char* ws = (char*)d_ws;                         // needs ~37.6 MB scratch
  __bf16* xs   = (__bf16*)(ws + WS_XS);
  __bf16* wqp  = (__bf16*)(ws + WS_WQ);
  float*  s2   = (float*) (ws + WS_S2);
  float*  epi  = (float*) (ws + WS_EPI);
  float*  zpad = (float*) (ws + WS_ZPAD);

  k_modx <<<NB * IMH, 256, 0, stream>>>(x, style, xs);
  k_wprep<<<COUT,     512, 0, stream>>>(weight, wqp, s2, zpad);
  k_demod<<<COUT,     256, 0, stream>>>(style, s2, epi);
  k_conv <<<dim3(NPIX / 128, COUT / 128), 256, 0, stream>>>(
      xs, wqp, epi, noise, bias, nstr, (const __bf16*)zpad, out);
}
